// LSTMEncoder_20590073217309
// MI455X (gfx1250) — compile-verified, matrix operand order checked by reading, not run on hardware
//
#include <hip/hip_runtime.h>

typedef __bf16 bf16_t;
typedef __attribute__((ext_vector_type(16))) __bf16 v16bf;
typedef __attribute__((ext_vector_type(8)))  __bf16 v8bf;
typedef __attribute__((ext_vector_type(4)))  __bf16 v4bf;
typedef __attribute__((ext_vector_type(8)))  float  v8f;
typedef __attribute__((ext_vector_type(4)))  int    v4i;

#define B_   128
#define T_   512
#define I_   512
#define H_   1024
#define G_   4096   // 4*H
#define K_   1536   // I + H  (fused x-proj + h-proj GEMM depth)
#define TGT_ 512

// ---- CDNA5 async global->LDS path (guarded; fallback = vector copy) ----
#if defined(__has_builtin)
#if __has_builtin(__builtin_amdgcn_global_load_async_to_lds_b128) && \
    __has_builtin(__builtin_amdgcn_s_wait_asynccnt)
#define USE_ASYNC_LDS 1
#endif
#endif
#ifndef USE_ASYNC_LDS
#define USE_ASYNC_LDS 0
#endif

#if USE_ASYNC_LDS
typedef __attribute__((address_space(1))) v4i glb_v4i;   // global (device) int4
typedef __attribute__((address_space(3))) v4i lds_v4i;   // LDS int4
#endif

__device__ __forceinline__ float fast_sigmoid(float x) {
    return 1.0f / (1.0f + __expf(-x));
}
__device__ __forceinline__ float fast_tanh(float x) {
    float e = __expf(2.0f * x);          // saturates correctly at +-1
    return 1.0f - 2.0f / (e + 1.0f);
}

// Build a 16-element bf16 WMMA operand from the ISA's two contiguous 16B runs
__device__ __forceinline__ v16bf load_frag(const bf16_t* p) {
    v8bf lo = *(const v8bf*)p;
    v8bf hi = *(const v8bf*)(p + 16);
    return __builtin_shufflevector(lo, hi, 0,1,2,3,4,5,6,7,8,9,10,11,12,13,14,15);
}

// ---------------------------------------------------------------------------
// Prep: Wcat = bf16([W_ih | W_hh]) rows of K_=1536; Wlin -> bf16;
// bias = b_ih + b_hh; zero c and both h ping-pong buffers. Runs every launch.
// ---------------------------------------------------------------------------
__global__ void lstm_prep(const float* __restrict__ Wih, const float* __restrict__ Whh,
                          const float* __restrict__ bih, const float* __restrict__ bhh,
                          const float* __restrict__ Wlin,
                          bf16_t* __restrict__ Wcat, bf16_t* __restrict__ WlinB,
                          float* __restrict__ bias, float* __restrict__ cbuf,
                          bf16_t* __restrict__ hbufs)
{
    const int stride = gridDim.x * blockDim.x;
    const int tid = blockIdx.x * blockDim.x + threadIdx.x;
    for (int i = tid; i < G_ * K_; i += stride) {
        int n = i / K_, k = i - n * K_;
        float v = (k < I_) ? Wih[(size_t)n * I_ + k] : Whh[(size_t)n * H_ + (k - I_)];
        Wcat[i] = (bf16_t)v;
    }
    for (int i = tid; i < TGT_ * H_; i += stride) WlinB[i] = (bf16_t)Wlin[i];
    for (int i = tid; i < G_; i += stride)        bias[i]  = bih[i] + bhh[i];
    for (int i = tid; i < B_ * H_; i += stride)   cbuf[i]  = 0.0f;
    for (int i = tid; i < 2 * B_ * H_; i += stride) hbufs[i] = (bf16_t)0.0f;
}

// ---------------------------------------------------------------------------
// One LSTM time step, fully fused:
//   gates[128,4096] = [X_t | h_in] @ Wcat^T + bias   (bf16 WMMA, f32 acc)
//   c = sigm(f)*c + sigm(i)*tanh(g); h_out = sigm(o)*tanh(c)
// Grid (4,32): 32-batch x 32-hidden-col block tile. 4 waves; wave w owns
// gate w and computes a 2x2 grid of 16x16 WMMA tiles (A and B frags each
// reused twice -> halves LDS + L2 traffic per FLOP).
// ---------------------------------------------------------------------------
__global__ void __launch_bounds__(128)
lstm_step(const float* __restrict__ XW, const bf16_t* __restrict__ Wcat,
          const float* __restrict__ bias, float* __restrict__ cbuf,
          const bf16_t* __restrict__ h_in, bf16_t* __restrict__ h_out, int t)
{
    __shared__ __align__(16) bf16_t Alds[32 * K_];   // 96 KB; reused for gate tiles

    const int tid  = threadIdx.x;
    const int lane = tid & 31;
    const int wave = tid >> 5;            // gate: 0=i 1=f 2=g 3=o
    const int b0   = blockIdx.x * 32;     // batch tile
    const int j0   = blockIdx.y * 32;     // hidden-column tile

    // ---- Phase 1a: X_t rows (f32 -> bf16, VALU convert) ----
    for (int p = tid; p < 32 * (I_ / 4); p += 128) {     // 4096 quads
        int row = p >> 7;                                 // I_/4 == 128
        int kq  = (p & 127) * 4;
        const float4 x = *(const float4*)&XW[((size_t)(b0 + row) * T_ + t) * I_ + kq];
        v4bf q;
        q.x = (bf16_t)x.x; q.y = (bf16_t)x.y; q.z = (bf16_t)x.z; q.w = (bf16_t)x.w;
        *(v4bf*)&Alds[row * K_ + kq] = q;
    }
    // ---- Phase 1b: h rows (pure copy -> async global->LDS on CDNA5) ----
    for (int p = tid; p < 32 * (H_ / 8); p += 128) {     // 4096 x 16B chunks
        int row = p >> 7;                                 // H_/8 == 128
        int k8  = (p & 127) * 8;
        const bf16_t* g = &h_in[(size_t)(b0 + row) * H_ + k8];
        bf16_t*       l = &Alds[row * K_ + I_ + k8];
#if USE_ASYNC_LDS
        __builtin_amdgcn_global_load_async_to_lds_b128(
            (glb_v4i*)(v4i*)g, (lds_v4i*)(v4i*)l, 0, 0);
#else
        *(v8bf*)l = *(const v8bf*)g;
#endif
    }
#if USE_ASYNC_LDS
    __builtin_amdgcn_s_wait_asynccnt(0);
#endif
    __syncthreads();

    // ---- Phase 2: 2x2 tiles of 16x16, K=1536, bias-seeded accumulators ----
    const int rsel = lane & 15;                 // row/col selector
    const int khi  = (lane >= 16) ? 8 : 0;      // 16-bit operand K split
    const float bv0 = bias[wave * H_ + j0 + rsel];
    const float bv1 = bias[wave * H_ + j0 + 16 + rsel];
    v8f acc00, acc01, acc10, acc11;
    #pragma unroll
    for (int r = 0; r < 8; ++r) {
        acc00[r] = bv0; acc01[r] = bv1; acc10[r] = bv0; acc11[r] = bv1;
    }

    const bf16_t* __restrict__ Br0 = Wcat + (size_t)(wave * H_ + j0 + rsel) * K_;
    const bf16_t* __restrict__ Br1 = Wcat + (size_t)(wave * H_ + j0 + 16 + rsel) * K_;
    const bf16_t* __restrict__ Ar0 = Alds + rsel * K_;
    const bf16_t* __restrict__ Ar1 = Alds + (16 + rsel) * K_;

    #pragma unroll 2
    for (int kc = 0; kc < K_; kc += 32) {
        v16bf a0 = load_frag(Ar0 + kc + khi);
        v16bf a1 = load_frag(Ar1 + kc + khi);
        v16bf b0 = load_frag(Br0 + kc + khi);
        v16bf b1 = load_frag(Br1 + kc + khi);
        acc00 = __builtin_amdgcn_wmma_f32_16x16x32_bf16(false, a0, false, b0, (short)0, acc00, false, false);
        acc01 = __builtin_amdgcn_wmma_f32_16x16x32_bf16(false, a0, false, b1, (short)0, acc01, false, false);
        acc10 = __builtin_amdgcn_wmma_f32_16x16x32_bf16(false, a1, false, b0, (short)0, acc10, false, false);
        acc11 = __builtin_amdgcn_wmma_f32_16x16x32_bf16(false, a1, false, b1, (short)0, acc11, false, false);
    }

    __syncthreads();  // all waves done reading Alds; safe to overlay gate tiles

    // ---- Phase 3: exchange gate tiles via LDS, elementwise cell update ----
    float* Glds = (float*)Alds;                 // [4 gates][32 m][32 n] = 16 KB
    {
        const int m0 = (lane >= 16) ? 8 : 0;    // C/D layout: VGPR r -> M = r + m0
        #pragma unroll
        for (int r = 0; r < 8; ++r) {
            Glds[wave * 1024 + (m0 + r) * 32 + rsel]            = acc00[r];
            Glds[wave * 1024 + (m0 + r) * 32 + 16 + rsel]       = acc01[r];
            Glds[wave * 1024 + (16 + m0 + r) * 32 + rsel]       = acc10[r];
            Glds[wave * 1024 + (16 + m0 + r) * 32 + 16 + rsel]  = acc11[r];
        }
    }
    __syncthreads();

    #pragma unroll
    for (int e = 0; e < 8; ++e) {
        int idx = tid + e * 128;                // 0..1023 of the 32x32 tile
        int m = idx >> 5, n = idx & 31;
        float gi = fast_sigmoid(Glds[0 * 1024 + m * 32 + n]);
        float gf = fast_sigmoid(Glds[1 * 1024 + m * 32 + n]);
        float gg = fast_tanh  (Glds[2 * 1024 + m * 32 + n]);
        float go = fast_sigmoid(Glds[3 * 1024 + m * 32 + n]);
        size_t ci = (size_t)(b0 + m) * H_ + (j0 + n);
        float c = gf * cbuf[ci] + gi * gg;
        cbuf[ci] = c;
        h_out[ci] = (bf16_t)(go * fast_tanh(c));
    }
}

// ---------------------------------------------------------------------------
// Final linear: out[128,512] = h @ W_lin^T + b_lin  (bf16 WMMA, f32 out)
// Grid (8,8), 4 waves per block, wave w owns N-tile (by*4+w).
// ---------------------------------------------------------------------------
__global__ void __launch_bounds__(128)
lstm_linear(const bf16_t* __restrict__ hfin, const bf16_t* __restrict__ WlinB,
            const float* __restrict__ blin, float* __restrict__ out)
{
    const int tid  = threadIdx.x;
    const int lane = tid & 31;
    const int wave = tid >> 5;
    const int b0   = blockIdx.x * 16;
    const int j0   = (blockIdx.y * 4 + wave) * 16;
    const int rsel = lane & 15;
    const int khi  = (lane >= 16) ? 8 : 0;

    const float bval = blin[j0 + rsel];
    v8f acc;
    #pragma unroll
    for (int r = 0; r < 8; ++r) acc[r] = bval;

    const bf16_t* __restrict__ Arow = hfin  + (size_t)(b0 + rsel) * H_;
    const bf16_t* __restrict__ Brow = WlinB + (size_t)(j0 + rsel) * H_;

    #pragma unroll 4
    for (int kc = 0; kc < H_; kc += 32) {
        v16bf a = load_frag(Arow + kc + khi);
        v16bf b = load_frag(Brow + kc + khi);
        acc = __builtin_amdgcn_wmma_f32_16x16x32_bf16(
                  false, a, false, b, (short)0, acc, false, false);
    }

    const int m0 = (lane >= 16) ? 8 : 0;
    #pragma unroll
    for (int r = 0; r < 8; ++r)
        out[(size_t)(b0 + m0 + r) * TGT_ + (j0 + rsel)] = acc[r];
}

// ---------------------------------------------------------------------------
// Workspace layout (bytes, 256-aligned):
//   [0)          Wcat  bf16 4096x1536   = 12,582,912
//   [12582912)   WlinB bf16 512x1024    =  1,048,576
//   [13631488)   bias  f32  4096        =     16,384
//   [13647872)   c     f32  128x1024    =    524,288
//   [14172160)   h ping-pong bf16 2x128x1024 = 524,288
//   total ~14.7 MB
// ---------------------------------------------------------------------------
extern "C" void kernel_launch(void* const* d_in, const int* in_sizes, int n_in,
                              void* d_out, int out_size, void* d_ws, size_t ws_size,
                              hipStream_t stream)
{
    (void)in_sizes; (void)n_in; (void)out_size; (void)ws_size;
    const float* XW   = (const float*)d_in[0];
    const float* Wih  = (const float*)d_in[1];
    const float* Whh  = (const float*)d_in[2];
    const float* bih  = (const float*)d_in[3];
    const float* bhh  = (const float*)d_in[4];
    const float* Wlin = (const float*)d_in[5];
    const float* blin = (const float*)d_in[6];
    float* out = (float*)d_out;

    char* ws = (char*)d_ws;
    bf16_t* Wcat  = (bf16_t*)(ws + 0);
    bf16_t* WlinB = (bf16_t*)(ws + 12582912);
    float*  bias  = (float*) (ws + 13631488);
    float*  cbuf  = (float*) (ws + 13647872);
    bf16_t* hb    = (bf16_t*)(ws + 14172160);
    bf16_t* h0 = hb;
    bf16_t* h1 = hb + (size_t)B_ * H_;

    lstm_prep<<<2048, 256, 0, stream>>>(Wih, Whh, bih, bhh, Wlin,
                                        Wcat, WlinB, bias, cbuf, hb);

    for (int s = 0; s < T_; ++s) {
        int t = T_ - 1 - s;                       // reversed time order
        const bf16_t* hin = (s & 1) ? h1 : h0;
        bf16_t*       hout = (s & 1) ? h0 : h1;
        lstm_step<<<dim3(B_ / 32, H_ / 32), 128, 0, stream>>>(
            XW, Wcat, bias, cbuf, hin, hout, t);
    }
    // T_=512 (even): last step (s=511, odd) wrote h0
    lstm_linear<<<dim3(B_ / 16, TGT_ / 64), 128, 0, stream>>>(h0, WlinB, blin, out);
}
